// SubjectBlockSA_21586505630382
// MI455X (gfx1250) — matrix operand order repaired and checked by reading, not compiled, not run on hardware
//
#include <hip/hip_runtime.h>

typedef __attribute__((ext_vector_type(16))) __bf16 v16bf;
typedef __attribute__((ext_vector_type(4)))  __bf16 bf16x4;
typedef __attribute__((ext_vector_type(8)))  float  v8f;
typedef __attribute__((ext_vector_type(4)))  float  f32x4;
typedef __attribute__((ext_vector_type(4)))  unsigned int u32x4;

// Problem sizes
constexpr int Bn = 512, Cc = 270, Tt = 360, Ss = 4, D1 = 270, KK = 1024, Ff = 2048;
// Padded dims for the folded matrix M (rows mult of 64, K mult of 32)
constexpr int DP = 320, CP = 288;
constexpr float TWO_PI = 6.28318530717958647692f;

// Workspace layout (bytes)
constexpr size_t OFF_PHIB = 0;                                   // phi in bf16 [S,C,F]
constexpr size_t SZ_PHIB  = (size_t)Ss * Cc * Ff * 2;            // 4,423,680
constexpr size_t OFF_ZB   = OFF_PHIB + SZ_PHIB;                  // Z in bf16 [S,D1,F]
constexpr size_t SZ_ZB    = (size_t)Ss * D1 * Ff * 2;            // 4,423,680
constexpr size_t OFF_SC   = OFF_ZB + SZ_ZB;                      // scores / softmax A (f32)
constexpr size_t SZ_SC    = (size_t)Ss * D1 * Cc * 4;            // 1,166,400
constexpr size_t OFF_T1   = OFF_SC + SZ_SC;                      // Wf @ Ws (f32)
constexpr size_t SZ_T1    = (size_t)Ss * D1 * D1 * 4;
constexpr size_t OFF_CV   = OFF_T1 + SZ_T1;                      // folded bias (f32)
constexpr size_t SZ_CV    = (size_t)Ss * D1 * 4;
constexpr size_t OFF_MB   = (OFF_CV + SZ_CV + 255) & ~(size_t)255; // M bf16 padded

static __device__ __forceinline__ unsigned short f2bf(float f) {
  unsigned int u = __float_as_uint(f);
  unsigned int r = (u + 0x7FFFu + ((u >> 16) & 1u)) >> 16;   // RNE
  return (unsigned short)r;
}

// ---- K1: Fourier features phi[s,c,:] = [cos, sin] -> bf16 ----
__global__ void phi_kernel(const float* __restrict__ loc, __bf16* __restrict__ phib) {
  const int c = blockIdx.x, s = blockIdx.y;
  const float x = loc[(s * Cc + c) * 2 + 0];
  const float y = loc[(s * Cc + c) * 2 + 1];
  __bf16* row = phib + ((size_t)s * Cc + c) * Ff;
  for (int p = threadIdx.x; p < KK; p += blockDim.x) {
    const int k = p >> 5, l = p & 31;
    const float ph = TWO_PI * (k * x + l * y);
    float sv, cvv;
    __sincosf(ph, &sv, &cvv);
    row[p] = (__bf16)cvv;
    row[KK + p] = (__bf16)sv;
  }
}

// ---- K1b: Z fp32 -> bf16 (vectorized) ----
__global__ void zb_kernel(const float* __restrict__ Z, __bf16* __restrict__ Zb) {
  const int idx = blockIdx.x * blockDim.x + threadIdx.x;   // over S*D1*F/4
  const f32x4 v = ((const f32x4*)Z)[idx];
  bf16x4 p;
  #pragma unroll
  for (int i = 0; i < 4; ++i) p[i] = (__bf16)v[i];
  ((bf16x4*)Zb)[idx] = p;
}

// ---- K2: scores[s,d,c] = <Z[s,d,:], phi[s,c,:]> via WMMA bf16 ----
// Both operands are K(f)-contiguous -> A and B fragments load DIRECTLY from
// global bf16 (no LDS). One wave per 16x16 tile, 64 K-chunks of 32.
__global__ __launch_bounds__(256) void scores_wmma(const __bf16* __restrict__ Zb,
                                                   const __bf16* __restrict__ phib,
                                                   float* __restrict__ sc) {
  constexpr int NT = 17;                         // ceil(270/16) tiles per dim
  const int s   = blockIdx.y;
  const int wid = threadIdx.x >> 5;
  const int lane = threadIdx.x & 31;
  const int w = blockIdx.x * 8 + wid;            // wave-uniform
  if (w >= NT * NT) return;
  const int dt = w / NT, ct = w % NT;
  const int halfSel = (lane < 16) ? 0 : 1;
  const int lr = lane & 15;
  // Row clamp: duplicated rows only feed outputs that are never stored.
  int dRow = dt * 16 + lr; dRow = (dRow < D1) ? dRow : (D1 - 1);
  int cRow = ct * 16 + lr; cRow = (cRow < Cc) ? cRow : (Cc - 1);
  // A (16x32): lane<16 bytes [0..15]&[32..47]; lane>=16 +16B
  const unsigned char* aB =
      (const unsigned char*)(Zb + ((size_t)s * D1 + dRow) * Ff) + (halfSel ? 16 : 0);
  // B (32x16): lane<16 bytes [0..15]&[32..47]; lane>=16 +16B
  const unsigned char* bB =
      (const unsigned char*)(phib + ((size_t)s * Cc + cRow) * Ff) + (halfSel ? 16 : 0);

  union Frag { u32x4 q[2]; v16bf v; };
  v8f acc = {0.f, 0.f, 0.f, 0.f, 0.f, 0.f, 0.f, 0.f};
  #pragma unroll 4
  for (int kc = 0; kc < Ff; kc += 32) {
    Frag a, bf_;
    const unsigned char* ap = aB + kc * 2;
    a.q[0]   = *(const u32x4*)(ap);
    a.q[1]   = *(const u32x4*)(ap + 32);
    const unsigned char* bp = bB + kc * 2;
    bf_.q[0] = *(const u32x4*)(bp);
    bf_.q[1] = *(const u32x4*)(bp + 32);
    acc = __builtin_amdgcn_wmma_f32_16x16x32_bf16(
        false, a.v, false, bf_.v, (short)0, acc, false, false);
  }
  #pragma unroll
  for (int r = 0; r < 8; ++r) {
    const int d = dt * 16 + r + (halfSel ? 8 : 0);
    const int c = ct * 16 + lr;
    if (d < D1 && c < Cc)
      sc[((size_t)s * D1 + d) * Cc + c] = acc[r];
  }
}

// ---- K3: softmax over c (one wave32 per row), in-place ----
__global__ void softmax_kernel(float* __restrict__ sc) {
  float* r = sc + (size_t)blockIdx.x * Cc;
  const int lane = threadIdx.x;
  float m = -3.4e38f;
  for (int c = lane; c < Cc; c += 32) m = fmaxf(m, r[c]);
  for (int o = 16; o > 0; o >>= 1) m = fmaxf(m, __shfl_xor(m, o, 32));
  float sum = 0.f;
  for (int c = lane; c < Cc; c += 32) { float e = __expf(r[c] - m); r[c] = e; sum += e; }
  for (int o = 16; o > 0; o >>= 1) sum += __shfl_xor(sum, o, 32);
  const float inv = 1.0f / sum;
  for (int c = lane; c < Cc; c += 32) r[c] *= inv;
}

// ---- K4: T1[s] = Wf @ Ws[s] ----
__global__ void t1_kernel(const float* __restrict__ Wf, const float* __restrict__ Wsub,
                          float* __restrict__ T1) {
  const int s = blockIdx.y;
  const int idx = blockIdx.x * blockDim.x + threadIdx.x;
  if (idx >= D1 * D1) return;
  const int i = idx / D1, j = idx % D1;
  const float* wrow = Wf + (size_t)i * D1;
  const float* wcol = Wsub + (size_t)s * D1 * D1 + j;
  float acc = 0.f;
  for (int e = 0; e < D1; ++e) acc += wrow[e] * wcol[(size_t)e * D1];
  T1[(size_t)s * D1 * D1 + idx] = acc;
}

// ---- K5: M[s] = T1[s] @ A[s] -> bf16 padded [DP x CP] (zero pad) ----
__global__ void mbf_kernel(const float* __restrict__ T1, const float* __restrict__ A,
                           unsigned short* __restrict__ Mb) {
  const int s = blockIdx.y;
  const int idx = blockIdx.x * blockDim.x + threadIdx.x;
  if (idx >= DP * CP) return;
  const int d = idx / CP, c = idx % CP;
  float acc = 0.f;
  if (d < D1 && c < Cc) {
    const float* trow = T1 + ((size_t)s * D1 + d) * D1;
    const float* acol = A + (size_t)s * D1 * Cc + c;
    for (int e = 0; e < D1; ++e) acc += trow[e] * acol[(size_t)e * Cc];
  }
  Mb[(size_t)s * DP * CP + idx] = f2bf(acc);
}

// ---- K6: cvec[s] = Wf @ bs[s] + bf ----
__global__ void cvec_kernel(const float* __restrict__ Wf, const float* __restrict__ bsub,
                            const float* __restrict__ bfin, float* __restrict__ cv) {
  const int idx = blockIdx.x * blockDim.x + threadIdx.x;
  if (idx >= Ss * D1) return;
  const int s = idx / D1, d = idx % D1;
  const float* wrow = Wf + (size_t)d * D1;
  const float* bvec = bsub + (size_t)s * D1;
  float acc = bfin[d];
  for (int e = 0; e < D1; ++e) acc += wrow[e] * bvec[e];
  cv[idx] = acc;
}

// ============================ Main GEMM ============================
// out[b] = M[s] @ X[b] + cvec[s]  via v_wmma_f32_16x16x32_bf16.
// WG = 256 threads = 8 waves as 4(row strips) x 2(col halves) on a 64x128 tile.
constexpr int LDB = 80;  // bytes per t-row in LDS B tile (64B data + 16B pad)

// Branchless X staging: thread loads a 4(c) x 4(t) fp32 block (4x global_load_b128,
// coalesced across lanes), transposes in-register, converts to bf16, writes
// 4x ds_store_b64 into the K-major [t][k] LDS tile.
// t-OOB: address CLAMP (padding t-columns are never stored -> dup data harmless).
// c-OOB (must be exact zero, feeds the K sum): only in the peeled last chunk.
template <bool CG>
static __device__ __forceinline__ void stageX(const float* __restrict__ Xb,
                                              unsigned char* __restrict__ ldsB,
                                              int kc, int t0, int tid) {
  const int tgrp = tid & 31;          // 32 t-groups of 4
  const int cgrp = tid >> 5;          // 8 c-groups of 4
  const int tl0  = tgrp * 4;          // local t base, 0..124
  int tg0 = t0 + tl0;
  tg0 = (tg0 < Tt) ? tg0 : (Tt - 4);  // clamp: Tt % 4 == 0, always fully in-bounds
  const int c0 = kc + cgrp * 4;
  const float* p0 = Xb + (size_t)c0 * Tt + tg0;

  f32x4 r[4];
  #pragma unroll
  for (int i = 0; i < 4; ++i) {
    const int cg = c0 + i;
    const float* p = CG ? (Xb + (size_t)((cg < Cc) ? cg : (Cc - 1)) * Tt + tg0)
                        : (p0 + (size_t)i * Tt);
    f32x4 v = *(const f32x4*)p;
    if (CG && cg >= Cc) v = (f32x4){0.f, 0.f, 0.f, 0.f};     // exact zero in K pad
    r[i] = v;
  }
  #pragma unroll
  for (int tt = 0; tt < 4; ++tt) {
    bf16x4 pk;
    #pragma unroll
    for (int i = 0; i < 4; ++i) pk[i] = (__bf16)r[i][tt];
    *(bf16x4*)(ldsB + (size_t)(tl0 + tt) * LDB + cgrp * 8) = pk;  // ds_store_b64
  }
}

static __device__ __forceinline__ void mmaStep(const unsigned char* __restrict__ aBase,
                                               const unsigned char* __restrict__ bBase,
                                               int kc, v8f* __restrict__ acc) {
  union Frag { u32x4 q[2]; v16bf v; };
  Frag a;
  const unsigned char* ap = aBase + kc * 2;
  a.q[0] = *(const u32x4*)(ap);
  a.q[1] = *(const u32x4*)(ap + 32);
  #pragma unroll
  for (int j = 0; j < 4; ++j) {
    const unsigned char* bp = bBase + (size_t)j * 16 * LDB;
    Frag bf_;
    bf_.q[0] = *(const u32x4*)(bp);
    bf_.q[1] = *(const u32x4*)(bp + 32);
    acc[j] = __builtin_amdgcn_wmma_f32_16x16x32_bf16(
        false, a.v, false, bf_.v, (short)0, acc[j], false, false);
  }
}

__global__ __launch_bounds__(256) void main_gemm(
    const float* __restrict__ X, const int* __restrict__ sidx,
    const unsigned short* __restrict__ Mb, const float* __restrict__ cv,
    float* __restrict__ out) {
  __shared__ __align__(16) unsigned char ldsB[128 * LDB];

  const int b       = blockIdx.z;
  const int rowTile = blockIdx.y;               // 0..4 -> d0 = 64*rowTile
  const int tTile   = blockIdx.x;               // 0..2 -> t0 = 128*tTile
  const int s       = sidx[b];

  const int tid  = threadIdx.x;
  const int lane = tid & 31;
  const int wid  = tid >> 5;
  const int wr   = wid >> 1;                    // 0..3 row strip (16 rows)
  const int wc   = wid & 1;                     // 0..1 col half  (64 cols)
  const int d0   = rowTile * 64;
  const int t0   = tTile * 128;
  const int halfSel = (lane < 16) ? 0 : 1;

  const unsigned short* Ms = Mb + (size_t)s * DP * CP;
  const float* Xb = X + (size_t)b * Cc * Tt;

  v8f acc[4];
  const v8f zero = {0.f, 0.f, 0.f, 0.f, 0.f, 0.f, 0.f, 0.f};
  #pragma unroll
  for (int j = 0; j < 4; ++j) acc[j] = zero;

  // A-operand: 16x32 bf16 strip from padded global (always in-bounds).
  const int aRow = d0 + wr * 16 + (lane & 15);
  const unsigned char* aBase =
      (const unsigned char*)(Ms + (size_t)aRow * CP) + (halfSel ? 16 : 0);
  // B fragment LDS base (per lane, constant across chunks and j)
  unsigned char* bBase = ldsB + (size_t)(wc * 64 + (lane & 15)) * LDB + (halfSel ? 16 : 0);

  // Steady-state chunks (no c-guard anywhere), last chunk peeled.
  #pragma unroll 1
  for (int kc = 0; kc < CP - 32; kc += 32) {
    __syncthreads();
    stageX<false>(Xb, ldsB, kc, t0, tid);
    __syncthreads();
    mmaStep(aBase, bBase, kc, acc);
  }
  __syncthreads();
  stageX<true>(Xb, ldsB, CP - 32, t0, tid);
  __syncthreads();
  mmaStep(aBase, bBase, CP - 32, acc);

  // Epilogue: C/D layout -> lane<16: M=r, N=lane; lane>=16: M=8+r, N=lane-16
  #pragma unroll
  for (int r = 0; r < 8; ++r) {
    const int m = r + (halfSel ? 8 : 0);
    const int d = d0 + wr * 16 + m;
    if (d >= D1) continue;
    const float bias = cv[s * D1 + d];
    #pragma unroll
    for (int j = 0; j < 4; ++j) {
      const int t = t0 + wc * 64 + j * 16 + (lane & 15);
      if (t < Tt)
        out[(size_t)b * D1 * Tt + (size_t)d * Tt + t] = acc[j][r] + bias;
    }
  }
}

extern "C" void kernel_launch(void* const* d_in, const int* in_sizes, int n_in,
                              void* d_out, int out_size, void* d_ws, size_t ws_size,
                              hipStream_t stream) {
  const float* X    = (const float*)d_in[0];
  const float* loc  = (const float*)d_in[1];
  const int*   sidx = (const int*)d_in[2];
  const float* Z    = (const float*)d_in[3];
  const float* Wsub = (const float*)d_in[4];
  const float* bsub = (const float*)d_in[5];
  const float* Wf   = (const float*)d_in[6];
  const float* bfin = (const float*)d_in[7];
  float* out = (float*)d_out;

  char* ws = (char*)d_ws;
  __bf16*         phib = (__bf16*)(ws + OFF_PHIB);
  __bf16*         Zb   = (__bf16*)(ws + OFF_ZB);
  float*          sc   = (float*)(ws + OFF_SC);
  float*          T1   = (float*)(ws + OFF_T1);
  float*          cv   = (float*)(ws + OFF_CV);
  unsigned short* Mb   = (unsigned short*)(ws + OFF_MB);

  phi_kernel    <<<dim3(Cc, Ss), 256, 0, stream>>>(loc, phib);
  zb_kernel     <<<(Ss * D1 * Ff / 4) / 256, 256, 0, stream>>>(Z, Zb);
  scores_wmma   <<<dim3((17 * 17 + 7) / 8, Ss), 256, 0, stream>>>(Zb, phib, sc);
  softmax_kernel<<<Ss * D1, 32, 0, stream>>>(sc);
  t1_kernel     <<<dim3((D1 * D1 + 255) / 256, Ss), 256, 0, stream>>>(Wf, Wsub, T1);
  mbf_kernel    <<<dim3((DP * CP + 255) / 256, Ss), 256, 0, stream>>>(T1, sc, Mb);
  cvec_kernel   <<<(Ss * D1 + 255) / 256, 256, 0, stream>>>(Wf, bsub, bfin, cv);
  main_gemm     <<<dim3(3, 5, Bn), 256, 0, stream>>>(X, sidx, Mb, cv, out);
}
